// MultiHeadAttention_89275190215527
// MI455X (gfx1250) — matrix operand order repaired and checked by reading, not compiled, not run on hardware
//
#include <hip/hip_runtime.h>

// ---------------------------------------------------------------------------
// MHA forward for gfx1250 (CDNA5), bf16 WMMA pipeline + TDM async tile loads.
// Shapes: B=2, S=2048, D=1024, H=16, Hd=64.
// ---------------------------------------------------------------------------

typedef __bf16 bf16;
typedef __attribute__((ext_vector_type(16))) __bf16 v16bf;
typedef __attribute__((ext_vector_type(8)))  __bf16 v8bf;
typedef __attribute__((ext_vector_type(8)))  float  v8f;
typedef __attribute__((ext_vector_type(4)))  unsigned int v4u;
typedef __attribute__((ext_vector_type(8)))  unsigned int v8u;

#define WMMA_BF16(A, B, C) \
  __builtin_amdgcn_wmma_f32_16x16x32_bf16(false, (A), false, (B), (short)0, (C), false, false)

// Combine two contiguous 8-bf16 chunks into a 16-element fragment register.
__device__ __forceinline__ v16bf frag_combine(v8bf lo, v8bf hi) {
  v16bf r;
#pragma unroll
  for (int i = 0; i < 8; ++i) { r[i] = lo[i]; r[i + 8] = hi[i]; }
  return r;
}

__device__ __forceinline__ v16bf load_frag16(const bf16* p0, const bf16* p1) {
  return frag_combine(*(const v8bf*)p0, *(const v8bf*)p1);
}

// A-fragment (16x32 bf16) from an f32 row (kB16 = 0 lanes 0-15, 16 lanes 16-31).
__device__ __forceinline__ v16bf load_afrag_f32(const float* row, int k0, int kB16) {
  v16bf r;
#pragma unroll
  for (int e = 0; e < 8; ++e) {
    r[e]     = (bf16)row[k0 + kB16 + e];
    r[e + 8] = (bf16)row[k0 + kB16 + 8 + e];
  }
  return r;
}

// ---------------------------------------------------------------------------
// TDM: 1D contiguous tile load (nbytes, multiple of 4) global -> LDS.
// Builds the D# descriptor per CDNA5 ISA and
// issues TENSOR_LOAD_TO_LDS (2 SGPR groups; VADDR2/3 = NULL => up-to-2D).
// Must be executed by exactly one wave; tracked with TENSORcnt.
// ---------------------------------------------------------------------------
__device__ __forceinline__ void tdm_load_1d(const void* lds_dst, const void* gsrc,
                                            unsigned nbytes) {
  const unsigned elems = nbytes >> 2;                 // 4-byte elements
  const unsigned lds_off = (unsigned)(unsigned long long)lds_dst;  // low 32 bits = LDS offset
  const unsigned long long ga = (unsigned long long)gsrc;

  v4u g0;
  g0[0] = 1u;                                         // count=1, user descriptor
  g0[1] = lds_off;                                    // lds_addr (bytes)
  g0[2] = (unsigned)ga;                               // global_addr[31:0]
  g0[3] = (unsigned)(ga >> 32) | (2u << 30);          // global_addr[56:32] | type=2

  v8u g1;
  g1[0] = (2u << 16);                                 // wg_mask=0, data_size=2 (4B)
  g1[1] = (elems & 0xffffu) << 16;                    // tensor_dim0[15:0]
  g1[2] = (elems >> 16) | (1u << 16);                 // tensor_dim0[31:16] | tensor_dim1=1
  g1[3] = (elems & 0xffffu) << 16;                    // tensor_dim1 hi=0 | tile_dim0
  g1[4] = 1u;                                         // tile_dim1=1, tile_dim2=0
  g1[5] = elems;                                      // tensor_dim0_stride[31:0]
  g1[6] = 0u;                                         // stride0 hi | stride1 lo
  g1[7] = 0u;                                         // stride1 hi
  asm volatile("tensor_load_to_lds %0, %1" :: "s"(g0), "s"(g1) : "memory");
}

// ---------------------------------------------------------------------------
// Kernel 1: QKV projection.  out[b][h][s][hd] = (x @ W + bias) * alpha  (bf16)
// grid: x = 16 (N block == head), y = 64, z = {Q,K,V} ; block 128 (4 waves).
// ---------------------------------------------------------------------------
__global__ __launch_bounds__(128) void qkv_proj_kernel(
    const float* __restrict__ x,
    const float* __restrict__ Wq, const float* __restrict__ bq,
    const float* __restrict__ Wk, const float* __restrict__ bk,
    const float* __restrict__ Wv, const float* __restrict__ bv,
    bf16* __restrict__ Qws, bf16* __restrict__ Kws, bf16* __restrict__ Vws) {
  const int z = blockIdx.z;
  const float* W    = (z == 0) ? Wq : (z == 1) ? Wk : Wv;
  const float* bias = (z == 0) ? bq : (z == 1) ? bk : bv;
  bf16* outp        = (z == 0) ? Qws : (z == 1) ? Kws : Vws;
  const float alpha = (z == 0) ? 0.125f : 1.0f;  // fold 1/sqrt(64) into Q

  __shared__ bf16 Wt[64][32];  // W tile transposed: [n][k]

  const int tid  = threadIdx.x;
  const int wv_  = tid >> 5;
  const int lane = tid & 31;
  const int ln   = lane & 15;
  const int half = lane >> 4;
  const int kB8 = half * 8, kB16 = half * 16, mB = half * 8;

  const int nb = blockIdx.x;
  const int mb = blockIdx.y;
  const int arow = mb * 64 + wv_ * 16 + ln;
  const float* xrow = x + (size_t)arow * 1024;

  v8f acc[4] = {};
  for (int kt = 0; kt < 32; ++kt) {
    const int k0 = kt * 32;
    for (int i = tid; i < 2048; i += 128) {
      const int kk = i >> 6, nn = i & 63;
      Wt[nn][kk] = (bf16)W[(size_t)(k0 + kk) * 1024 + nb * 64 + nn];
    }
    __syncthreads();

    const v16bf a = load_afrag_f32(xrow, k0, kB16);
    v16bf bf[4];
#pragma unroll
    for (int nt = 0; nt < 4; ++nt)
      bf[nt] = load_frag16(&Wt[nt * 16 + ln][kB16], &Wt[nt * 16 + ln][kB16 + 8]);
#pragma unroll
    for (int nt = 0; nt < 4; ++nt) acc[nt] = WMMA_BF16(a, bf[nt], acc[nt]);
    __syncthreads();
  }

  const int rbase = mb * 64 + wv_ * 16 + mB;
#pragma unroll
  for (int nt = 0; nt < 4; ++nt) {
    const int n  = nb * 64 + nt * 16 + ln;
    const int hd = nt * 16 + ln;
    const float bv_ = bias[n];
#pragma unroll
    for (int e = 0; e < 8; ++e) {
      const int r = rbase + e;
      const int b = r >> 11, s = r & 2047;
      const float v = (acc[nt][e] + bv_) * alpha;
      outp[(((size_t)(b * 16 + nb) * 2048) + s) * 64 + hd] = (bf16)v;
    }
  }
}

// ---------------------------------------------------------------------------
// Kernel 2: flash attention forward with double-buffered TDM tile loads.
// grid: x = S/64 = 32, y = H = 16, z = B = 2 ; block = 128 (4 waves).
// K tile and V tile are contiguous 4 KB blocks -> trivial 1D descriptors.
// Wave 0 issues TENSOR_LOAD_TO_LDS for tile kt+1 while all waves compute kt.
// ---------------------------------------------------------------------------
__global__ __launch_bounds__(128) void attn_fwd_kernel(
    const bf16* __restrict__ Qws, const bf16* __restrict__ Kws,
    const bf16* __restrict__ Vws, bf16* __restrict__ ctxws,
    float* __restrict__ mws, float* __restrict__ lws) {
  const int b = blockIdx.z, h = blockIdx.y, qb = blockIdx.x;
  const int bh = b * 16 + h;

  __shared__ bf16 Kl[2][32][64];    // K tiles, natural [key][hd], double buffered
  __shared__ bf16 Vs[2][32][64];    // V staging, natural, double buffered
  __shared__ bf16 Vt[64][32];       // V tile transposed [hd][key]
  __shared__ bf16 Pb[4][16][32];    // per-wave P scratch

  const int tid  = threadIdx.x;
  const int wv_  = tid >> 5;
  const int lane = tid & 31;
  const int ln   = lane & 15;
  const int half = lane >> 4;
  const int kB8 = half * 8, kB16 = half * 16, mB = half * 8;

  const int qrow = qb * 64 + wv_ * 16 + ln;
  const bf16* Qrow = Qws + ((size_t)bh * 2048 + qrow) * 64;
  const v16bf qa0 = load_frag16(Qrow + kB16,      Qrow + kB16 + 8);
  const v16bf qa1 = load_frag16(Qrow + 32 + kB16, Qrow + 32 + kB16 + 8);

  const bf16* Kbase = Kws + (size_t)bh * 2048 * 64;
  const bf16* Vbase = Vws + (size_t)bh * 2048 * 64;

  float m_[8], l_[8];
  v8f acc[4] = {};
#pragma unroll
  for (int e = 0; e < 8; ++e) { m_[e] = -1e30f; l_[e] = 0.0f; }

  // Prologue: async-load tile 0 into buffer 0.
  if (wv_ == 0) {
    tdm_load_1d(&Kl[0][0][0], Kbase, 4096);
    tdm_load_1d(&Vs[0][0][0], Vbase, 4096);
  }

  for (int kt = 0; kt < 64; ++kt) {
    const int cur = kt & 1;
    __syncthreads();  // (0) everyone done with buffer cur^1 -> safe to refill
    if (wv_ == 0) {
      if (kt + 1 < 64) {
        tdm_load_1d(&Kl[cur ^ 1][0][0], Kbase + (size_t)(kt + 1) * 32 * 64, 4096);
        tdm_load_1d(&Vs[cur ^ 1][0][0], Vbase + (size_t)(kt + 1) * 32 * 64, 4096);
        __builtin_amdgcn_s_wait_tensorcnt(2);  // oldest pair (tile kt) complete
      } else {
        __builtin_amdgcn_s_wait_tensorcnt(0);
      }
    }
    __syncthreads();  // (1) tile kt resident in LDS for all waves

    // Transpose V staging -> Vt (LDS -> LDS).
    {
      const int r = tid >> 2, c0 = (tid & 3) * 16;
      const v8bf a0 = *(const v8bf*)&Vs[cur][r][c0];
      const v8bf a1 = *(const v8bf*)&Vs[cur][r][c0 + 8];
#pragma unroll
      for (int i = 0; i < 8; ++i) {
        Vt[c0 + i][r]     = a0[i];
        Vt[c0 + 8 + i][r] = a1[i];
      }
    }
    __syncthreads();  // (2) Vt ready

    // Scores: 16q x 32key.
    v8f s0 = {}, s1 = {};
    {
      const v16bf b00 = load_frag16(&Kl[cur][ln][kB16],      &Kl[cur][ln][kB16 + 8]);
      const v16bf b01 = load_frag16(&Kl[cur][ln][32 + kB16], &Kl[cur][ln][32 + kB16 + 8]);
      const v16bf b10 = load_frag16(&Kl[cur][16 + ln][kB16],      &Kl[cur][16 + ln][kB16 + 8]);
      const v16bf b11 = load_frag16(&Kl[cur][16 + ln][32 + kB16], &Kl[cur][16 + ln][32 + kB16 + 8]);
      s0 = WMMA_BF16(qa0, b00, s0);
      s0 = WMMA_BF16(qa1, b01, s0);
      s1 = WMMA_BF16(qa0, b10, s1);
      s1 = WMMA_BF16(qa1, b11, s1);
    }

    // Online softmax (rows in element index, 16 key cols per lane-half).
    float mn[8], corr[8];
#pragma unroll
    for (int e = 0; e < 8; ++e) {
      float t = fmaxf(s0[e], s1[e]);
      t = fmaxf(t, __shfl_xor(t, 1, 32));
      t = fmaxf(t, __shfl_xor(t, 2, 32));
      t = fmaxf(t, __shfl_xor(t, 4, 32));
      t = fmaxf(t, __shfl_xor(t, 8, 32));
      mn[e]   = fmaxf(m_[e], t);
      corr[e] = __expf(m_[e] - mn[e]);
    }
    float p0[8], p1[8];
#pragma unroll
    for (int e = 0; e < 8; ++e) {
      p0[e] = __expf(s0[e] - mn[e]);
      p1[e] = __expf(s1[e] - mn[e]);
      float t = p0[e] + p1[e];
      t += __shfl_xor(t, 1, 32);
      t += __shfl_xor(t, 2, 32);
      t += __shfl_xor(t, 4, 32);
      t += __shfl_xor(t, 8, 32);
      l_[e] = l_[e] * corr[e] + t;
      m_[e] = mn[e];
    }
#pragma unroll
    for (int nt = 0; nt < 4; ++nt)
#pragma unroll
      for (int e = 0; e < 8; ++e) acc[nt][e] *= corr[e];

    // C-layout -> A-layout for P via per-wave LDS round trip.
#pragma unroll
    for (int e = 0; e < 8; ++e) {
      Pb[wv_][mB + e][ln]      = (bf16)p0[e];
      Pb[wv_][mB + e][16 + ln] = (bf16)p1[e];
    }
    asm volatile("s_wait_dscnt 0" ::: "memory");  // own-wave LDS stores visible

    const v16bf pa = load_frag16(&Pb[wv_][ln][kB16], &Pb[wv_][ln][kB16 + 8]);
    v16bf vb[4];
#pragma unroll
    for (int nt = 0; nt < 4; ++nt)
      vb[nt] = load_frag16(&Vt[nt * 16 + ln][kB16], &Vt[nt * 16 + ln][kB16 + 8]);
#pragma unroll
    for (int nt = 0; nt < 4; ++nt) acc[nt] = WMMA_BF16(pa, vb[nt], acc[nt]);
  }

  // Epilogue: normalize, store ctx (bf16) + stats (m, 1/l).
  float invl[8];
#pragma unroll
  for (int e = 0; e < 8; ++e) invl[e] = 1.0f / l_[e];

  const int rb = qb * 64 + wv_ * 16 + mB;
  bf16* ctxb = ctxws + (size_t)bh * 2048 * 64;
#pragma unroll
  for (int nt = 0; nt < 4; ++nt) {
    const int hd = nt * 16 + ln;
#pragma unroll
    for (int e = 0; e < 8; ++e)
      ctxb[(size_t)(rb + e) * 64 + hd] = (bf16)(acc[nt][e] * invl[e]);
  }
  if (ln == 0) {
#pragma unroll
    for (int e = 0; e < 8; ++e) {
      mws[(size_t)bh * 2048 + rb + e] = m_[e];
      lws[(size_t)bh * 2048 + rb + e] = invl[e];
    }
  }
}

// ---------------------------------------------------------------------------
// Kernel 3: attn_mean[b][q][k] = (1/H) * sum_h exp(score - m) * (1/l).
// ---------------------------------------------------------------------------
__global__ __launch_bounds__(128) void attn_mean_kernel(
    const bf16* __restrict__ Qws, const bf16* __restrict__ Kws,
    const float* __restrict__ mws, const float* __restrict__ lws,
    float* __restrict__ attn_out) {
  const int b  = blockIdx.z;
  const int qt = blockIdx.y;
  const int tid  = threadIdx.x;
  const int wv_  = tid >> 5;
  const int lane = tid & 31;
  const int ln   = lane & 15;
  const int half = lane >> 4;
  const int kB8 = half * 8, kB16 = half * 16, mB = half * 8;

  const int kcb = (blockIdx.x * 4 + wv_) * 16;

  v8f acc = {};
  for (int h = 0; h < 16; ++h) {
    const int bh = b * 16 + h;
    const bf16* Qrow = Qws + ((size_t)bh * 2048 + qt * 16 + ln) * 64;
    const v16bf qa0 = load_frag16(Qrow + kB16,      Qrow + kB16 + 8);
    const v16bf qa1 = load_frag16(Qrow + 32 + kB16, Qrow + 32 + kB16 + 8);
    const bf16* Krow = Kws + ((size_t)bh * 2048 + kcb + ln) * 64;
    const v16bf kb0 = load_frag16(Krow + kB16,      Krow + kB16 + 8);
    const v16bf kb1 = load_frag16(Krow + 32 + kB16, Krow + 32 + kB16 + 8);
    v8f s = {};
    s = WMMA_BF16(qa0, kb0, s);
    s = WMMA_BF16(qa1, kb1, s);
    const float* mp = mws + (size_t)bh * 2048 + qt * 16 + mB;
    const float* lp = lws + (size_t)bh * 2048 + qt * 16 + mB;  // holds 1/l
#pragma unroll
    for (int e = 0; e < 8; ++e) acc[e] += __expf(s[e] - mp[e]) * lp[e];
  }

  float* outb = attn_out + (size_t)b * 2048 * 2048;
#pragma unroll
  for (int e = 0; e < 8; ++e)
    outb[(size_t)(qt * 16 + mB + e) * 2048 + kcb + ln] = acc[e] * 0.0625f;
}

// ---------------------------------------------------------------------------
// Kernel 4: output projection. out = ctx(reordered) @ Wo + bo  (f32 out)
// ---------------------------------------------------------------------------
__global__ __launch_bounds__(128) void out_proj_kernel(
    const bf16* __restrict__ ctxws, const float* __restrict__ Wo,
    const float* __restrict__ bo, float* __restrict__ out) {
  __shared__ bf16 Wt[64][32];

  const int tid  = threadIdx.x;
  const int wv_  = tid >> 5;
  const int lane = tid & 31;
  const int ln   = lane & 15;
  const int half = lane >> 4;
  const int kB8 = half * 8, kB16 = half * 16, mB = half * 8;

  const int nb = blockIdx.x;
  const int mb = blockIdx.y;
  const int arow = mb * 64 + wv_ * 16 + ln;
  const int ab = arow >> 11, as = arow & 2047;

  v8f acc[4] = {};
  for (int kt = 0; kt < 32; ++kt) {
    const int k0 = kt * 32;
    for (int i = tid; i < 2048; i += 128) {
      const int kk = i >> 6, nn = i & 63;
      Wt[nn][kk] = (bf16)Wo[(size_t)(k0 + kk) * 1024 + nb * 64 + nn];
    }
    __syncthreads();

    const int hh = k0 >> 6, hd0 = k0 & 63;
    const bf16* arowp = ctxws + (((size_t)(ab * 16 + hh) * 2048) + as) * 64;
    const v16bf a = load_frag16(arowp + hd0 + kB16, arowp + hd0 + kB16 + 8);
    v16bf bf[4];
#pragma unroll
    for (int nt = 0; nt < 4; ++nt)
      bf[nt] = load_frag16(&Wt[nt * 16 + ln][kB16], &Wt[nt * 16 + ln][kB16 + 8]);
#pragma unroll
    for (int nt = 0; nt < 4; ++nt) acc[nt] = WMMA_BF16(a, bf[nt], acc[nt]);
    __syncthreads();
  }

  const int rbase = mb * 64 + wv_ * 16 + mB;
#pragma unroll
  for (int nt = 0; nt < 4; ++nt) {
    const int n = nb * 64 + nt * 16 + ln;
    const float bv_ = bo[n];
#pragma unroll
    for (int e = 0; e < 8; ++e)
      out[(size_t)(rbase + e) * 1024 + n] = acc[nt][e] + bv_;
  }
}

// ---------------------------------------------------------------------------
// Launch
// ---------------------------------------------------------------------------
extern "C" void kernel_launch(void* const* d_in, const int* in_sizes, int n_in,
                              void* d_out, int out_size, void* d_ws, size_t ws_size,
                              hipStream_t stream) {
  (void)in_sizes; (void)n_in; (void)out_size; (void)ws_size;
  const float* x  = (const float*)d_in[0];
  const float* Wq = (const float*)d_in[1];
  const float* bq = (const float*)d_in[2];
  const float* Wk = (const float*)d_in[3];
  const float* bk = (const float*)d_in[4];
  const float* Wv = (const float*)d_in[5];
  const float* bv = (const float*)d_in[6];
  const float* Wo = (const float*)d_in[7];
  const float* bo = (const float*)d_in[8];

  float* out = (float*)d_out;
  float* attn_out = out + (size_t)2 * 2048 * 1024;

  char* ws = (char*)d_ws;
  bf16* Qws   = (bf16*)(ws);                                   // 8 MB
  bf16* Kws   = (bf16*)(ws + ((size_t)8 << 20));               // 8 MB
  bf16* Vws   = (bf16*)(ws + ((size_t)16 << 20));              // 8 MB
  bf16* ctxws = (bf16*)(ws + ((size_t)24 << 20));              // 8 MB
  float* mws  = (float*)(ws + ((size_t)32 << 20));             // 256 KB
  float* lws  = (float*)(ws + ((size_t)32 << 20) + (256u << 10));

  qkv_proj_kernel<<<dim3(16, 64, 3), 128, 0, stream>>>(x, Wq, bq, Wk, bk, Wv, bv,
                                                       Qws, Kws, Vws);
  attn_fwd_kernel<<<dim3(32, 16, 2), 128, 0, stream>>>(Qws, Kws, Vws, ctxws, mws, lws);
  attn_mean_kernel<<<dim3(32, 128, 2), 128, 0, stream>>>(Qws, Kws, mws, lws, attn_out);
  out_proj_kernel<<<dim3(16, 64, 1), 128, 0, stream>>>(ctxws, Wo, bo, out);
}
